// Decoder_21715354650008
// MI455X (gfx1250) — matrix operand order repaired and checked by reading, not compiled, not run on hardware
//
#include <hip/hip_runtime.h>
#include <hip/hip_bf16.h>
#include <stdint.h>

typedef __attribute__((ext_vector_type(16))) __bf16 v16bf;
typedef __attribute__((ext_vector_type(8)))  float  v8f;

#define TSTEPS 51
#define BATCH  64
#define PLOC   196
#define ENCD   2048
#define AD     512
#define DD     512
#define VOC    10000
#define LCAP   52
#define XK     3072   // emb(512) + ctx(2048) + h(512)

// ---------- helpers ----------
__device__ __forceinline__ unsigned short f2bf(float f) {
  unsigned int u = __float_as_uint(f);
  u += 0x7FFFu + ((u >> 16) & 1u);          // round-to-nearest-even
  return (unsigned short)(u >> 16);
}

// A-fragment (16x32 bf16, wave32) element index per ISA 7.12.2:
// lane = (within16>>3)*16 + (m&15); vgpr v = (kr>>4)*4 + ((kr&7)>>1); pair bit = k&1
__device__ __forceinline__ size_t pack_a_idx(int m, int k, int K) {
  int mtile = m >> 4, ks = k >> 5, kr = k & 31;
  int w16 = kr & 15;
  int v = (kr >> 4) * 4 + ((w16 & 7) >> 1);
  int lane = (w16 >> 3) * 16 + (m & 15);
  return ((((size_t)mtile * (K >> 5) + ks) * 32 + lane) * 8 + v) * 2 + (w16 & 1);
}

// ---------- sort (argsort of -lengths, stable) ----------
__global__ void k_sort(const int* __restrict__ caplen, const int* __restrict__ caps,
                       int* __restrict__ sort_ws, int* __restrict__ dlen_ws,
                       int* __restrict__ caps_ws,
                       float* __restrict__ out_caps, float* __restrict__ out_dlen,
                       float* __restrict__ out_sort) {
  __shared__ int len[BATCH];
  __shared__ int sidx[BATCH];
  int tid = threadIdx.x;
  len[tid] = caplen[tid];
  __syncthreads();
  int r = 0;
  for (int j = 0; j < BATCH; ++j) {
    int lj = len[j];
    r += (lj > len[tid]) || (lj == len[tid] && j < tid);
  }
  sidx[r] = tid;
  __syncthreads();
  int s = sidx[tid];
  sort_ws[tid] = s;
  out_sort[tid] = (float)s;
  int dl = len[s] - 1;
  dlen_ws[tid] = dl;
  out_dlen[tid] = (float)dl;
  for (int j = 0; j < LCAP; ++j) {
    int cv = caps[s * LCAP + j];
    caps_ws[tid * LCAP + j] = cv;
    out_caps[tid * LCAP + j] = (float)cv;
  }
}

// ---------- bias fold: b_ih + b_hh ----------
__global__ void k_bias_sum(const float* __restrict__ a, const float* __restrict__ b,
                           float* __restrict__ o, int n) {
  int i = blockIdx.x * blockDim.x + threadIdx.x;
  if (i < n) o[i] = a[i] + b[i];
}

// ---------- pack f32 weight [Ksrc x N] row-major into B-fragment bf16 layout ----------
// B (32x16 bf16): lane = ((kr&15)>>3)*16 + n%16 ; vgpr v = (kr>>4)*4 + ((kr&7)>>1) ; pair bit = k&1
__global__ void k_pack_b(const float* __restrict__ src, uint16_t* __restrict__ dst,
                         int Ksrc, int N, int Ktot, int koff) {
  size_t i = (size_t)blockIdx.x * blockDim.x + threadIdx.x;
  if (i >= (size_t)Ksrc * N) return;
  int ksrc = (int)(i / N);
  int n = (int)(i % N);
  int k = ksrc + koff;
  int ks = k >> 5, kr = k & 31;
  int within = kr & 15;
  size_t dw = (((size_t)(n >> 4) * (Ktot >> 5) + ks) * 32 + ((within >> 3) * 16 + (n & 15))) * 8
              + (kr >> 4) * 4 + ((within & 7) >> 1);
  dst[dw * 2 + (within & 1)] = f2bf(src[(size_t)ksrc * N + n]);
}

// ---------- mean over P of sorted encoder ----------
__global__ void k_mean(const float* __restrict__ enc, const int* __restrict__ sort_ws,
                       float* __restrict__ mean) {
  int i = blockIdx.x * blockDim.x + threadIdx.x;   // BATCH*ENCD
  int b = i >> 11, k = i & (ENCD - 1);
  const float* p = enc + ((size_t)sort_ws[b] * PLOC) * ENCD + k;
  float s = 0.f;
  for (int j = 0; j < PLOC; ++j) s += p[(size_t)j * ENCD];
  mean[i] = s * (1.0f / PLOC);
}

// ---------- initial h,c = mean_enc @ W + b ----------
__global__ void k_init(const float* __restrict__ mean,
                       const float* __restrict__ Wh, const float* __restrict__ bh,
                       const float* __restrict__ Wc, const float* __restrict__ bc,
                       float* __restrict__ h, float* __restrict__ c,
                       uint16_t* __restrict__ h_pack) {
  int i = blockIdx.x * blockDim.x + threadIdx.x;   // BATCH*DD
  int b = i >> 9, d = i & (DD - 1);
  const float* m = mean + (size_t)b * ENCD;
  float sh = bh[d], sc = bc[d];
  for (int k = 0; k < ENCD; ++k) {
    float mv = m[k];
    sh += mv * Wh[(size_t)k * DD + d];
    sc += mv * Wc[(size_t)k * DD + d];
  }
  h[i] = sh;
  c[i] = sc;
  h_pack[pack_a_idx(b, d, DD)] = f2bf(sh);
}

// ---------- sorted encoder -> packed bf16 A for the att1 GEMM ----------
__global__ void k_enc_pack(const float* __restrict__ enc, const int* __restrict__ sort_ws,
                           uint16_t* __restrict__ dst) {
  size_t i = (size_t)blockIdx.x * blockDim.x + threadIdx.x;
  if (i >= (size_t)BATCH * PLOC * ENCD) return;
  int k = (int)(i & (ENCD - 1));
  int m = (int)(i >> 11);             // b*196+p
  int b = m / PLOC, p = m % PLOC;
  float v = enc[((size_t)sort_ws[b] * PLOC + p) * ENCD + k];
  dst[pack_a_idx(m, k, ENCD)] = f2bf(v);
}

// ---------- bf16 WMMA GEMM: C[MxN] = Apack[MxK] * Bpack[KxN] (+bias, mask) ----------
// block = 128 threads (4 waves); wave w -> M-tile (blockIdx.y*4+w); 4 N-tiles per wave.
// Branch-free K-loop: out-of-range N-tiles are clamped (recompute last tile, results
// discarded by the uniform epilogue guard) so accumulators never cross control flow.
__global__ __launch_bounds__(128) void k_gemm(
    const uint32_t* __restrict__ Apack,
    const uint32_t* __restrict__ Bpack,
    float* __restrict__ C,
    const float* __restrict__ bias,
    const int* __restrict__ dlen,
    int N, int K, int ldc, int t) {
  const int lane = threadIdx.x & 31;
  const int wave = threadIdx.x >> 5;
  const int Ks = K >> 5;
  const int mtile = blockIdx.y * 4 + wave;
  const int ntile0 = blockIdx.x * 4;
  const int ntiles = N >> 4;

  int nt0 = ntile0 + 0; nt0 = nt0 < ntiles ? nt0 : ntiles - 1;
  int nt1 = ntile0 + 1; nt1 = nt1 < ntiles ? nt1 : ntiles - 1;
  int nt2 = ntile0 + 2; nt2 = nt2 < ntiles ? nt2 : ntiles - 1;
  int nt3 = ntile0 + 3; nt3 = nt3 < ntiles ? nt3 : ntiles - 1;

  const uint32_t* ap  = Apack + (((size_t)mtile * Ks) * 32 + lane) * 8;
  const uint32_t* bp0 = Bpack + (((size_t)nt0 * Ks) * 32 + lane) * 8;
  const uint32_t* bp1 = Bpack + (((size_t)nt1 * Ks) * 32 + lane) * 8;
  const uint32_t* bp2 = Bpack + (((size_t)nt2 * Ks) * 32 + lane) * 8;
  const uint32_t* bp3 = Bpack + (((size_t)nt3 * Ks) * 32 + lane) * 8;

  v8f acc0 = {0.f, 0.f, 0.f, 0.f, 0.f, 0.f, 0.f, 0.f};
  v8f acc1 = acc0, acc2 = acc0, acc3 = acc0;

  for (int ks = 0; ks < Ks; ++ks) {
    v16bf a  = *(const v16bf*)ap;  ap  += 256;
    v16bf b0 = *(const v16bf*)bp0; bp0 += 256;
    v16bf b1 = *(const v16bf*)bp1; bp1 += 256;
    v16bf b2 = *(const v16bf*)bp2; bp2 += 256;
    v16bf b3 = *(const v16bf*)bp3; bp3 += 256;
    acc0 = __builtin_amdgcn_wmma_f32_16x16x32_bf16(false, a, false, b0, (short)0, acc0, false, false);
    acc1 = __builtin_amdgcn_wmma_f32_16x16x32_bf16(false, a, false, b1, (short)0, acc1, false, false);
    acc2 = __builtin_amdgcn_wmma_f32_16x16x32_bf16(false, a, false, b2, (short)0, acc2, false, false);
    acc3 = __builtin_amdgcn_wmma_f32_16x16x32_bf16(false, a, false, b3, (short)0, acc3, false, false);
  }

  const int nlo = lane & 15;
  const int mrow0 = mtile * 16 + ((lane >> 4) << 3);
  v8f accs[4] = {acc0, acc1, acc2, acc3};
  #pragma unroll
  for (int j = 0; j < 4; ++j) {
    int nt = ntile0 + j;
    if (nt >= ntiles) continue;              // uniform per-wave guard, epilogue only
    int n = nt * 16 + nlo;
    float bz = bias ? bias[n] : 0.f;
    #pragma unroll
    for (int r = 0; r < 8; ++r) {
      int m = mrow0 + r;
      float val = accs[j][r] + bz;
      if (dlen) val = (dlen[m] > t) ? val : 0.f;
      C[(size_t)m * ldc + n] = val;
    }
  }
}

// ---------- per-step attention: e, softmax, ctx, gate; builds packed x=[emb|g*ctx|h] ----------
__global__ void k_attn(const float* __restrict__ att1, const float* __restrict__ att2,
                       const float* __restrict__ wfull, const float* __restrict__ bfull,
                       const float* __restrict__ enc, const int* __restrict__ sort_ws,
                       const int* __restrict__ dlen_ws,
                       const float* __restrict__ gatepre, const float* __restrict__ h,
                       const int* __restrict__ caps_ws, const float* __restrict__ embt,
                       uint16_t* __restrict__ x_pack, float* __restrict__ out_alpha, int t) {
  int b = blockIdx.x, tid = threadIdx.x;
  __shared__ float sh[PLOC];
  __shared__ float red[256];

  if (tid < PLOC) {
    const float* a1 = att1 + ((size_t)b * PLOC + tid) * AD;
    const float* a2 = att2 + (size_t)b * AD;
    float e = bfull[0];
    for (int a = 0; a < AD; ++a) {
      float v = a1[a] + a2[a];
      v = v > 0.f ? v : 0.f;
      e += v * wfull[a];
    }
    sh[tid] = e;
  }
  __syncthreads();
  red[tid] = (tid < PLOC) ? sh[tid] : -3.0e38f;
  __syncthreads();
  for (int s = 128; s > 0; s >>= 1) {
    if (tid < s) red[tid] = fmaxf(red[tid], red[tid + s]);
    __syncthreads();
  }
  float mx = red[0];
  __syncthreads();
  float ex = 0.f;
  if (tid < PLOC) ex = __expf(sh[tid] - mx);
  red[tid] = (tid < PLOC) ? ex : 0.f;
  __syncthreads();
  for (int s = 128; s > 0; s >>= 1) {
    if (tid < s) red[tid] += red[tid + s];
    __syncthreads();
  }
  float inv = 1.f / red[0];
  __syncthreads();
  float mk = (dlen_ws[b] > t) ? 1.f : 0.f;
  if (tid < PLOC) {
    float al = ex * inv;
    sh[tid] = al;
    out_alpha[((size_t)b * TSTEPS + t) * PLOC + tid] = al * mk;
  }
  __syncthreads();

  // x[0:512) = embedding of current token (packed bf16)
  int tok = caps_ws[b * LCAP + t];
  for (int k = tid; k < DD; k += 256)
    x_pack[pack_a_idx(b, k, XK)] = f2bf(embt[(size_t)tok * DD + k]);

  // x[512:2560) = sigmoid(gatepre) * (alpha . enc)
  const float* eb = enc + (size_t)sort_ws[b] * PLOC * ENCD;
  for (int e = tid; e < ENCD; e += 256) {
    float s = 0.f;
    for (int p = 0; p < PLOC; ++p) s += sh[p] * eb[(size_t)p * ENCD + e];
    float gate = 1.f / (1.f + __expf(-gatepre[(size_t)b * ENCD + e]));
    x_pack[pack_a_idx(b, DD + e, XK)] = f2bf(gate * s);
  }

  // x[2560:3072) = h_t  (fuses the h@W_hh matmul into the big GEMM)
  for (int k = tid; k < DD; k += 256)
    x_pack[pack_a_idx(b, 2560 + k, XK)] = f2bf(h[(size_t)b * DD + k]);
}

// ---------- LSTM cell ----------
__global__ void k_lstm(const float* __restrict__ gates, float* __restrict__ c,
                       float* __restrict__ h, uint16_t* __restrict__ h_pack) {
  int i = blockIdx.x * blockDim.x + threadIdx.x;  // BATCH*DD
  int b = i >> 9, d = i & (DD - 1);
  const float* g = gates + (size_t)b * 4 * DD;
  float gi = g[d], gf = g[DD + d], gg = g[2 * DD + d], go = g[3 * DD + d];
  float si = 1.f / (1.f + __expf(-gi));
  float sf = 1.f / (1.f + __expf(-gf));
  float so = 1.f / (1.f + __expf(-go));
  float cn = sf * c[i] + si * tanhf(gg);
  float hn = so * tanhf(cn);
  c[i] = cn;
  h[i] = hn;
  h_pack[pack_a_idx(b, d, DD)] = f2bf(hn);
}

extern "C" void kernel_launch(void* const* d_in, const int* in_sizes, int n_in,
                              void* d_out, int out_size, void* d_ws, size_t ws_size,
                              hipStream_t stream) {
  (void)in_sizes; (void)n_in; (void)out_size; (void)ws_size;
  const float* enc    = (const float*)d_in[0];
  const int*   caps   = (const int*)d_in[1];
  const int*   caplen = (const int*)d_in[2];
  const float* Wenc   = (const float*)d_in[3];
  const float* benc   = (const float*)d_in[4];
  const float* Wdec   = (const float*)d_in[5];
  const float* bdec   = (const float*)d_in[6];
  const float* wfull  = (const float*)d_in[7];
  const float* bfull  = (const float*)d_in[8];
  const float* embt   = (const float*)d_in[9];
  const float* Wih    = (const float*)d_in[10];
  const float* bih    = (const float*)d_in[11];
  const float* Whh    = (const float*)d_in[12];
  const float* bhh    = (const float*)d_in[13];
  const float* WhI    = (const float*)d_in[14];
  const float* bhI    = (const float*)d_in[15];
  const float* WcI    = (const float*)d_in[16];
  const float* bcI    = (const float*)d_in[17];
  const float* Wbeta  = (const float*)d_in[18];
  const float* bbeta  = (const float*)d_in[19];
  const float* Wfc    = (const float*)d_in[20];
  const float* bfc    = (const float*)d_in[21];

  float* out = (float*)d_out;
  size_t o_caps  = (size_t)BATCH * TSTEPS * VOC;
  size_t o_dlen  = o_caps + (size_t)BATCH * LCAP;
  size_t o_alpha = o_dlen + BATCH;
  size_t o_sort  = o_alpha + (size_t)BATCH * TSTEPS * PLOC;

  char* wsp = (char*)d_ws;
  auto alloc = [&](size_t bytes) -> void* {
    void* p = (void*)wsp;
    wsp += (bytes + 255) & ~(size_t)255;
    return p;
  };
  int*      sort_ws = (int*)alloc(BATCH * 4);
  int*      dlen_ws = (int*)alloc(BATCH * 4);
  int*      caps_ws = (int*)alloc((size_t)BATCH * LCAP * 4);
  float*    bsum    = (float*)alloc((size_t)4 * DD * 4);
  float*    h       = (float*)alloc((size_t)BATCH * DD * 4);
  float*    c       = (float*)alloc((size_t)BATCH * DD * 4);
  float*    mean    = (float*)alloc((size_t)BATCH * ENCD * 4);
  float*    att2    = (float*)alloc((size_t)BATCH * AD * 4);
  float*    gpre    = (float*)alloc((size_t)BATCH * ENCD * 4);
  float*    gates   = (float*)alloc((size_t)BATCH * 4 * DD * 4);
  float*    att1    = (float*)alloc((size_t)BATCH * PLOC * AD * 4);
  uint16_t* h_pack  = (uint16_t*)alloc((size_t)BATCH * DD * 2);
  uint16_t* x_pack  = (uint16_t*)alloc((size_t)BATCH * XK * 2);
  uint16_t* pWenc   = (uint16_t*)alloc((size_t)ENCD * AD * 2);
  uint16_t* pWdec   = (uint16_t*)alloc((size_t)DD * AD * 2);
  uint16_t* pWbeta  = (uint16_t*)alloc((size_t)DD * ENCD * 2);
  uint16_t* pWcomb  = (uint16_t*)alloc((size_t)XK * 2048 * 2);
  uint16_t* pWfc    = (uint16_t*)alloc((size_t)DD * VOC * 2);
  uint16_t* pEnc    = (uint16_t*)alloc((size_t)BATCH * PLOC * ENCD * 2);

  k_sort<<<1, BATCH, 0, stream>>>(caplen, caps, sort_ws, dlen_ws, caps_ws,
                                  out + o_caps, out + o_dlen, out + o_sort);
  k_bias_sum<<<(4 * DD + 255) / 256, 256, 0, stream>>>(bih, bhh, bsum, 4 * DD);

  auto packb = [&](const float* src, uint16_t* dst, int Ksrc, int Nn, int Ktot, int koff) {
    size_t tot = (size_t)Ksrc * Nn;
    k_pack_b<<<(unsigned)((tot + 255) / 256), 256, 0, stream>>>(src, dst, Ksrc, Nn, Ktot, koff);
  };
  packb(Wenc, pWenc, ENCD, AD, ENCD, 0);
  packb(Wdec, pWdec, DD, AD, DD, 0);
  packb(Wbeta, pWbeta, DD, ENCD, DD, 0);
  packb(Wih, pWcomb, 2560, 2048, XK, 0);
  packb(Whh, pWcomb, DD, 2048, XK, 2560);
  packb(Wfc, pWfc, DD, VOC, DD, 0);

  k_mean<<<(BATCH * ENCD) / 256, 256, 0, stream>>>(enc, sort_ws, mean);
  k_init<<<(BATCH * DD) / 256, 256, 0, stream>>>(mean, WhI, bhI, WcI, bcI, h, c, h_pack);
  {
    size_t tot = (size_t)BATCH * PLOC * ENCD;
    k_enc_pack<<<(unsigned)((tot + 255) / 256), 256, 0, stream>>>(enc, sort_ws, pEnc);
  }
  // att1 = enc_sorted @ W_enc + b_enc : M=12544, N=512, K=2048
  k_gemm<<<dim3(8, 196), 128, 0, stream>>>((const uint32_t*)pEnc, (const uint32_t*)pWenc,
                                           att1, benc, nullptr, AD, ENCD, AD, 0);

  for (int t = 0; t < TSTEPS; ++t) {
    // att2 = h @ W_dec + b_dec  (64x512, K=512)
    k_gemm<<<dim3(8, 1), 128, 0, stream>>>((const uint32_t*)h_pack, (const uint32_t*)pWdec,
                                           att2, bdec, nullptr, AD, DD, AD, 0);
    // gatepre = h @ f_beta_W + f_beta_b  (64x2048, K=512)
    k_gemm<<<dim3(32, 1), 128, 0, stream>>>((const uint32_t*)h_pack, (const uint32_t*)pWbeta,
                                            gpre, bbeta, nullptr, ENCD, DD, ENCD, 0);
    // e / softmax / ctx / gate / build packed x
    k_attn<<<BATCH, 256, 0, stream>>>(att1, att2, wfull, bfull, enc, sort_ws, dlen_ws,
                                      gpre, h, caps_ws, embt, x_pack, out + o_alpha, t);
    // gates = x @ [W_ih;W_hh] + (b_ih+b_hh)  (64x2048, K=3072)
    k_gemm<<<dim3(32, 1), 128, 0, stream>>>((const uint32_t*)x_pack, (const uint32_t*)pWcomb,
                                            gates, bsum, nullptr, 2048, XK, 2048, 0);
    k_lstm<<<(BATCH * DD) / 256, 256, 0, stream>>>(gates, c, h, h_pack);
    // pred = h_new @ fc_W + fc_b, masked, scattered to out[b, t, :]  (64x10000, K=512)
    k_gemm<<<dim3(157, 1), 128, 0, stream>>>((const uint32_t*)h_pack, (const uint32_t*)pWfc,
                                             out + (size_t)t * VOC, bfc, dlen_ws,
                                             VOC, DD, TSTEPS * VOC, t);
  }
}
